// Qwen3MoeSparseMoeBlock_32916629356772
// MI455X (gfx1250) — matrix operand order repaired and checked by reading, not compiled, not run on hardware
//
#include <hip/hip_runtime.h>
#include <hip/hip_bf16.h>

#define T_TOK 1024
#define HDIM  2048
#define IDIM  768
#define NEXP  64
#define TOPK  8
#define CAP   1040            // per-expert list capacity (1024 + pad)
#define KC    256             // stage-1 K-chunk of H staged in LDS
#define XLD   (KC + 8)        // padded LDS row (bf16 elems) for X chunk
#define ALD   (IDIM + 8)      // padded LDS row (bf16 elems) for activations

typedef __bf16 bf16_t;
typedef __attribute__((ext_vector_type(16))) __bf16 v16bf;
typedef __attribute__((ext_vector_type(4)))  __bf16 bf16x4;
typedef __attribute__((ext_vector_type(8)))  float  v8f;

// ---------------------------------------------------------------------------
// WMMA helpers (CDNA5: D = A(16x32 bf16) x B(32x16 bf16) + C(16x16 f32))
// ---------------------------------------------------------------------------
__device__ __forceinline__ v8f wmma_bf16(v16bf a, v16bf b, v8f c) {
  return __builtin_amdgcn_wmma_f32_16x16x32_bf16(
      /*neg_a=*/false, a, /*neg_b=*/false, b,
      /*c_mod=*/(short)0, c, /*reuse_a=*/false, /*reuse_b=*/false);
}

// A-matrix fragment (16x32, MxK) from LDS bf16 row-major (ld elems).
// ISA 7.12.2: lanes 0-15 -> M=lane, K={0..7,16..23}; lanes 16-31 -> K={8..15,24..31}
__device__ __forceinline__ v16bf fragA_lds(const bf16_t* p, int ld) {
  const int lane = threadIdx.x & 31;
  const int r    = lane & 15;
  const int kb   = (lane >> 4) << 3;          // 0 or 8
  const bf16_t* q = p + r * ld + kb;
  v16bf f;
#pragma unroll
  for (int j = 0; j < 8; ++j) f[j]     = q[j];
#pragma unroll
  for (int j = 0; j < 8; ++j) f[8 + j] = q[16 + j];
  return f;
}

// B-matrix fragment (32x16, KxN) for D = X * W^T: B[k][n] = W[n][k], W rows
// contiguous along k. ISA: lane n holds column n; lanes 0-15 K={0..7,16..23},
// lanes 16-31 K={8..15,24..31} (same order as fragA_lds).
__device__ __forceinline__ v16bf fragB_gl(const float* p, int ld) {
  const int lane = threadIdx.x & 31;
  const int n    = lane & 15;
  const int kh   = (lane >> 4) << 3;          // 0 or 8
  const float* q = p + (size_t)n * ld + kh;
  v16bf f;
#pragma unroll
  for (int j = 0; j < 8; ++j) f[j] = (bf16_t)q[j];
#pragma unroll
  for (int j = 0; j < 8; ++j) f[8 + j] = (bf16_t)q[16 + j];
  return f;
}

// ---------------------------------------------------------------------------
// Kernel 1: router — logits, softmax, top-8, renormalize
// ---------------------------------------------------------------------------
__global__ void __launch_bounds__(256)
router_kernel(const float* __restrict__ x, const float* __restrict__ gate_w,
              int* __restrict__ topk_idx, float* __restrict__ topk_w) {
  __shared__ float sx[HDIM];
  __shared__ float slog[NEXP];
  const int t   = blockIdx.x;
  const int tid = threadIdx.x;
  const float* xr = x + (size_t)t * HDIM;
  for (int i = tid * 4; i < HDIM; i += 256 * 4)
    *reinterpret_cast<float4*>(sx + i) = *reinterpret_cast<const float4*>(xr + i);
  __syncthreads();

  const int wave = tid >> 5, lane = tid & 31;
#pragma unroll
  for (int ee = 0; ee < NEXP / 8; ++ee) {     // 8 waves x 8 experts
    const int e = wave * 8 + ee;
    const float* g = gate_w + (size_t)e * HDIM;
    float acc = 0.f;
    for (int h = lane * 4; h < HDIM; h += 128) {
      float4 gv = *reinterpret_cast<const float4*>(g + h);
      acc += gv.x * sx[h] + gv.y * sx[h + 1] + gv.z * sx[h + 2] + gv.w * sx[h + 3];
    }
#pragma unroll
    for (int off = 16; off > 0; off >>= 1) acc += __shfl_down(acc, off, 32);
    if (lane == 0) slog[e] = acc;
  }
  __syncthreads();

  if (tid == 0) {
    float m = slog[0];
    for (int e = 1; e < NEXP; ++e) m = fmaxf(m, slog[e]);
    float p[NEXP]; float s = 0.f;
    for (int e = 0; e < NEXP; ++e) { p[e] = __expf(slog[e] - m); s += p[e]; }
    const float inv = 1.f / s;
    for (int e = 0; e < NEXP; ++e) p[e] *= inv;
    int idx[TOPK]; float w[TOPK]; float wsum = 0.f;
    for (int k = 0; k < TOPK; ++k) {          // iterative argmax (ties: lowest idx)
      int best = 0; float bv = -1.f;
      for (int e = 0; e < NEXP; ++e) if (p[e] > bv) { bv = p[e]; best = e; }
      idx[k] = best; w[k] = bv; wsum += bv; p[best] = -2.f;
    }
    const float invw = 1.f / wsum;
    for (int k = 0; k < TOPK; ++k) {
      topk_idx[t * TOPK + k] = idx[k];
      topk_w  [t * TOPK + k] = w[k] * invw;
    }
  }
}

// ---------------------------------------------------------------------------
// Kernel 2: deterministic per-expert token lists (ascending token order)
// ---------------------------------------------------------------------------
__global__ void build_lists_kernel(const int* __restrict__ topk_idx,
                                   const float* __restrict__ topk_w,
                                   int* __restrict__ tok, float* __restrict__ wgt,
                                   int* __restrict__ slot, int* __restrict__ ntiles) {
  const int e = threadIdx.x;
  int*   te = tok  + e * CAP;
  float* we = wgt  + e * CAP;
  int*   se = slot + e * CAP;
  int cnt = 0;
  for (int i = 0; i < T_TOK * TOPK; ++i) {
    if (topk_idx[i] == e) { te[cnt] = i / TOPK; we[cnt] = topk_w[i]; se[cnt] = i; ++cnt; }
  }
  while (cnt & 15) { te[cnt] = 0; we[cnt] = 0.f; se[cnt] = -1; ++cnt; }   // pad tile
  ntiles[e] = cnt >> 4;
}

// ---------------------------------------------------------------------------
// Kernel 3: fused expert SwiGLU MLP. Block = 1 expert x 16-token tile.
// 512 threads = 16 waves. Stage 1: 48 I-tiles (3/wave) over H; SwiGLU;
// Stage 2: 128 H-tiles (8/wave) over I. bf16 WMMA, f32 accumulate.
// ---------------------------------------------------------------------------
__global__ void __launch_bounds__(512, 1)
expert_kernel(const float* __restrict__ x,
              const float* __restrict__ w_gate,
              const float* __restrict__ w_up,
              const float* __restrict__ w_down,
              const int* __restrict__ tok, const float* __restrict__ wgt,
              const int* __restrict__ slot, const int* __restrict__ ntiles,
              float* __restrict__ slotbuf) {
  const int e = blockIdx.x;
  if ((int)blockIdx.y >= ntiles[e]) return;

  __shared__ alignas(16) bf16_t lds_x[16 * XLD];     // 8.4 KB
  __shared__ alignas(16) bf16_t lds_act[16 * ALD];   // 24.8 KB
  __shared__ int   s_tok[16];
  __shared__ float s_w[16];
  __shared__ int   s_slot[16];

  const int tid  = threadIdx.x;
  const int wave = tid >> 5;

  if (tid < 16) {
    const int ent = e * CAP + blockIdx.y * 16 + tid;
    s_tok[tid]  = tok[ent];
    s_w[tid]    = wgt[ent];
    s_slot[tid] = slot[ent];
  }

  const v8f vzero = {0.f, 0.f, 0.f, 0.f, 0.f, 0.f, 0.f, 0.f};
  v8f accG[3], accU[3];
#pragma unroll
  for (int i = 0; i < 3; ++i) { accG[i] = vzero; accU[i] = vzero; }

  // ---- Stage 1: G = X*Wg^T, U = X*Wu^T, accumulated over H in KC chunks ----
  for (int kb = 0; kb < HDIM; kb += KC) {
    __syncthreads();                              // covers s_tok + lds_x reuse
    for (int idx = tid; idx < 16 * (KC / 4); idx += 512) {
      const int r  = idx >> 6;                    // KC/4 == 64 groups per row
      const int c4 = (idx & 63) << 2;
      const float4 v = *reinterpret_cast<const float4*>(
          x + (size_t)s_tok[r] * HDIM + kb + c4);
      bf16x4 b;
      b[0] = (bf16_t)v.x; b[1] = (bf16_t)v.y; b[2] = (bf16_t)v.z; b[3] = (bf16_t)v.w;
      *reinterpret_cast<bf16x4*>(lds_x + r * XLD + c4) = b;     // ds_store_b64
    }
    __syncthreads();

    for (int ks = 0; ks < KC / 32; ++ks) {
      const v16bf af = fragA_lds(lds_x + ks * 32, XLD);
      const size_t kgl = (size_t)kb + (size_t)ks * 32;
#pragma unroll
      for (int nt = 0; nt < 3; ++nt) {
        const int i0 = (wave * 3 + nt) << 4;      // I-tile base (0..767)
        const v16bf bg = fragB_gl(w_gate + ((size_t)e * IDIM + i0) * HDIM + kgl, HDIM);
        accG[nt] = wmma_bf16(af, bg, accG[nt]);
        const v16bf bu = fragB_gl(w_up   + ((size_t)e * IDIM + i0) * HDIM + kgl, HDIM);
        accU[nt] = wmma_bf16(af, bu, accU[nt]);
      }
    }
  }

  // ---- SwiGLU in registers -> bf16 activation tile in LDS ----
  {
    const int lane = tid & 31;
    const int n    = lane & 15;                   // i-col within tile
    const int mhi  = (lane >> 4) << 3;            // token row offset 0/8
#pragma unroll
    for (int nt = 0; nt < 3; ++nt) {
      const int i0 = (wave * 3 + nt) << 4;
#pragma unroll
      for (int v = 0; v < 8; ++v) {
        const float g = accG[nt][v], u = accU[nt][v];
        const float a = (g / (1.f + __expf(-g))) * u;   // silu(g)*u
        lds_act[(mhi + v) * ALD + i0 + n] = (bf16_t)a;
      }
    }
  }
  __syncthreads();

  // ---- Stage 2: D = A * Wd^T over I ----
  v8f accD[8];
#pragma unroll
  for (int i = 0; i < 8; ++i) accD[i] = vzero;

  for (int kt = 0; kt < IDIM / 32; ++kt) {        // 24 K-steps
    const v16bf af = fragA_lds(lds_act + kt * 32, ALD);
#pragma unroll
    for (int nt = 0; nt < 8; ++nt) {
      const int h0 = (wave * 8 + nt) << 4;        // H-tile base (0..2047)
      const v16bf bd = fragB_gl(w_down + ((size_t)e * HDIM + h0) * IDIM + kt * 32, IDIM);
      accD[nt] = wmma_bf16(af, bd, accD[nt]);
    }
  }

  // ---- Scale by routing weight, write (token,k) slot exactly once ----
  {
    const int lane = tid & 31;
    const int n    = lane & 15;
    const int mhi  = (lane >> 4) << 3;
#pragma unroll
    for (int nt = 0; nt < 8; ++nt) {
      const int h0 = (wave * 8 + nt) << 4;
#pragma unroll
      for (int v = 0; v < 8; ++v) {
        const int m  = mhi + v;
        const int sl = s_slot[m];
        if (sl >= 0)
          slotbuf[(size_t)sl * HDIM + h0 + n] = accD[nt][v] * s_w[m];
      }
    }
  }
}

// ---------------------------------------------------------------------------
// Kernel 4: deterministic combine over the 8 top-k slots per token
// ---------------------------------------------------------------------------
__global__ void __launch_bounds__(256)
combine_kernel(const float* __restrict__ slotbuf, float* __restrict__ out) {
  const size_t idx = (size_t)blockIdx.x * 256 + threadIdx.x;
  if (idx >= (size_t)T_TOK * HDIM) return;
  const size_t t = idx / HDIM, h = idx % HDIM;
  float s = 0.f;
#pragma unroll
  for (int k = 0; k < TOPK; ++k)
    s += slotbuf[(t * TOPK + k) * HDIM + h];
  out[idx] = s;
}

// ---------------------------------------------------------------------------
extern "C" void kernel_launch(void* const* d_in, const int* in_sizes, int n_in,
                              void* d_out, int out_size, void* d_ws, size_t ws_size,
                              hipStream_t stream) {
  (void)in_sizes; (void)n_in; (void)out_size; (void)ws_size;
  const float* x      = (const float*)d_in[0];   // [1, T, H]
  const float* gate_w = (const float*)d_in[1];   // [E, H]
  const float* wg     = (const float*)d_in[2];   // [E, I, H]
  const float* wu     = (const float*)d_in[3];   // [E, I, H]
  const float* wd     = (const float*)d_in[4];   // [E, H, I]
  float* out = (float*)d_out;

  char* ws = (char*)d_ws;
  size_t off = 0;
  auto alloc = [&](size_t bytes) -> void* {
    void* p = ws + off;
    off = (off + bytes + 255) & ~(size_t)255;
    return p;
  };
  int*   topk_idx = (int*)  alloc(sizeof(int)   * T_TOK * TOPK);
  float* topk_w   = (float*)alloc(sizeof(float) * T_TOK * TOPK);
  int*   tok      = (int*)  alloc(sizeof(int)   * NEXP * CAP);
  float* wgt      = (float*)alloc(sizeof(float) * NEXP * CAP);
  int*   slot     = (int*)  alloc(sizeof(int)   * NEXP * CAP);
  int*   ntiles   = (int*)  alloc(sizeof(int)   * NEXP);
  float* slotbuf  = (float*)alloc(sizeof(float) * (size_t)T_TOK * TOPK * HDIM); // 64 MB

  router_kernel<<<T_TOK, 256, 0, stream>>>(x, gate_w, topk_idx, topk_w);
  build_lists_kernel<<<1, NEXP, 0, stream>>>(topk_idx, topk_w, tok, wgt, slot, ntiles);
  expert_kernel<<<dim3(NEXP, T_TOK / 16), 512, 0, stream>>>(
      x, wg, wu, wd, tok, wgt, slot, ntiles, slotbuf);
  combine_kernel<<<(int)(((size_t)T_TOK * HDIM + 255) / 256), 256, 0, stream>>>(slotbuf, out);
}
